// BiLSTMCRF_55078660604442
// MI455X (gfx1250) — matrix operand order repaired and checked by reading, not compiled, not run on hardware
//
#include <hip/hip_runtime.h>
#include <hip/hip_bf16.h>

// ---------------------------------------------------------------------------
// BiLSTM-CRF on MI455X (gfx1250): f16 WMMA GEMMs + LDS-resident WMMA recurrence
// ---------------------------------------------------------------------------

typedef __attribute__((ext_vector_type(16))) _Float16 v16h;
typedef __attribute__((ext_vector_type(8)))  float    v8f;

#define Bn 512
#define Ln 200
#define En 256
#define Hn 128
#define M_TOT (Bn * Ln)   // 102400 token rows

// ---- WMMA fragment loaders (layouts per CDNA5 ISA 7.12.2, wave32) ----------

// A: 16x32 f16, row-major source with leading dim `ld` (halfs).
// lane l<16: row=l, K = 0..7 and 16..23 ; lane l>=16: row=l-16, K = 8..15 and 24..31
__device__ __forceinline__ v16h frag_a(const _Float16* base, int ld) {
  int l = threadIdx.x & 31;
  const _Float16* p = base + (l & 15) * ld + ((l >> 4) << 3);
  v16h a;
#pragma unroll
  for (int j = 0; j < 8; ++j) a[j] = p[j];
#pragma unroll
  for (int j = 0; j < 8; ++j) a[8 + j] = p[16 + j];
  return a;
}

// B: 32x16 (KxN) built from weight rows W[n][k] (row-major, leading dim `ld`).
// lane l<16: col=l, K = 0..7 and 16..23 ; lane l>=16: col=l-16, K = 8..15 and 24..31
__device__ __forceinline__ v16h frag_b(const _Float16* base, int ld) {
  int l = threadIdx.x & 31;
  const _Float16* p = base + (l & 15) * ld + ((l >> 4) << 3);
  v16h b;
#pragma unroll
  for (int j = 0; j < 8; ++j) b[j] = p[j];
#pragma unroll
  for (int j = 0; j < 8; ++j) b[8 + j] = p[16 + j];
  return b;
}

__device__ __forceinline__ v8f vzero8() {
  v8f z = {0.f, 0.f, 0.f, 0.f, 0.f, 0.f, 0.f, 0.f};
  return z;
}

__device__ __forceinline__ float sigm(float x) { return 1.0f / (1.0f + __expf(-x)); }
__device__ __forceinline__ float leaky(float x) { return x >= 0.f ? x : 0.01f * x; }

// ---- workspace layout (bytes) ----------------------------------------------
#define OFF_XH     ((size_t)0)                       // x f16        [102400,256]  52.4 MB
#define OFF_XG     ((size_t)52428800)                // xg f16       [102400,512] 104.9 MB (reused as h1)
#define OFF_HCAT   ((size_t)157286400)               // h concat f16 [102400,256]  52.4 MB
#define OFF_WH     ((size_t)209715200)               // f16 weights ~1.06 MB
#define OFF_BIAS   ((size_t)210771968)               // fused lstm biases f32 [1024]
#define OFF_LOGITS ((size_t)210776064)               // logits f32 [102400,6]
// f16 weight sub-offsets (in halfs)
#define WO_WIHF 0
#define WO_WIHB 131072
#define WO_WHHF 262144
#define WO_WHHB 327680
#define WO_W1   393216
#define WO_W2   458752
#define WO_WT   524288
#define W_TOTAL 528384

// ---- prep: cast weights f32->f16, pad Wt to 16 rows, fuse LSTM biases ------
__global__ void k_prep(const float* Wih_f, const float* Whh_f, const float* bih_f,
                       const float* bhh_f, const float* Wih_b, const float* Whh_b,
                       const float* bih_b, const float* bhh_b, const float* W1,
                       const float* W2, const float* Wt, _Float16* wh, float* biasf) {
  int idx = blockIdx.x * blockDim.x + threadIdx.x;
  int i = idx;
  if (i < 131072) { wh[WO_WIHF + i] = (_Float16)Wih_f[i]; return; }
  i -= 131072;
  if (i < 131072) { wh[WO_WIHB + i] = (_Float16)Wih_b[i]; return; }
  i -= 131072;
  if (i < 65536)  { wh[WO_WHHF + i] = (_Float16)Whh_f[i]; return; }
  i -= 65536;
  if (i < 65536)  { wh[WO_WHHB + i] = (_Float16)Whh_b[i]; return; }
  i -= 65536;
  if (i < 65536)  { wh[WO_W1 + i] = (_Float16)W1[i]; return; }
  i -= 65536;
  if (i < 65536)  { wh[WO_W2 + i] = (_Float16)W2[i]; return; }
  i -= 65536;
  if (i < 4096)   { int r = i >> 8, k = i & 255;
                    wh[WO_WT + i] = (_Float16)(r < 6 ? Wt[r * 256 + k] : 0.f); return; }
  i -= 4096;
  if (i < 512)    { biasf[i] = bih_f[i] + bhh_f[i]; return; }
  i -= 512;
  if (i < 512)    { biasf[512 + i] = bih_b[i] + bhh_b[i]; return; }
}

// ---- embedding gather + f32->f16 -------------------------------------------
__global__ void k_embed(const int* sent, const float* emb, _Float16* xh) {
  size_t row = blockIdx.x;
  int tok = sent[row];
  const float* src = emb + (size_t)tok * En;
  xh[row * En + threadIdx.x] = (_Float16)src[threadIdx.x];
}

// ---- generic WMMA GEMM: out[M,N] = act(A[M,K] * W[N,K]^T + bias) -----------
// block: 256 threads (8 waves), tile 64(M) x 128(N), K step 32
#define LDS_PAD 34
__global__ void k_gemm(const _Float16* __restrict__ A, const _Float16* __restrict__ W,
                       const float* __restrict__ bias, _Float16* __restrict__ out,
                       int lda, int ldc, int K, int act) {
  __shared__ _Float16 As[64 * LDS_PAD];
  __shared__ _Float16 Ws[128 * LDS_PAD];
  int tid = threadIdx.x;
  int wave = tid >> 5;
  int wm = wave & 1;        // 0..1 -> 32-row half
  int wn = wave >> 1;       // 0..3 -> 32-col slice
  size_t m0 = (size_t)blockIdx.y * 64;
  int n0 = blockIdx.x * 128;

  v8f acc[2][2];
#pragma unroll
  for (int i = 0; i < 2; ++i)
#pragma unroll
    for (int j = 0; j < 2; ++j) acc[i][j] = vzero8();

  int ar = tid >> 2, aseg = (tid & 3) << 3;   // A: 64 rows x 4 chunks of 8 halfs
  int wr = tid >> 1, wseg = (tid & 1) << 4;   // W: 128 rows x 2 chunks of 16 halfs

  for (int kk = 0; kk < K; kk += 32) {
    // stage A and W tiles into LDS
#pragma unroll
    for (int q = 0; q < 8; ++q)
      As[ar * LDS_PAD + aseg + q] = A[(m0 + ar) * lda + kk + aseg + q];
#pragma unroll
    for (int q = 0; q < 16; ++q)
      Ws[wr * LDS_PAD + wseg + q] = W[(size_t)(n0 + wr) * K + kk + wseg + q];
    if (kk + 32 < K) {  // gfx1250 global_prefetch_b8: locality 3 -> WGP scope,
                        // pulls next K tile through all cache levels
      __builtin_prefetch(&A[(m0 + ar) * lda + kk + 32 + aseg], 0, 3);
      __builtin_prefetch(&W[(size_t)(n0 + wr) * K + kk + 32 + wseg], 0, 3);
    }
    __syncthreads();

    v16h a0 = frag_a(&As[(wm * 32 + 0)  * LDS_PAD], LDS_PAD);
    v16h a1 = frag_a(&As[(wm * 32 + 16) * LDS_PAD], LDS_PAD);
    v16h b0 = frag_b(&Ws[(wn * 32 + 0)  * LDS_PAD], LDS_PAD);
    v16h b1 = frag_b(&Ws[(wn * 32 + 16) * LDS_PAD], LDS_PAD);
    acc[0][0] = __builtin_amdgcn_wmma_f32_16x16x32_f16(false, a0, false, b0, (short)0, acc[0][0], false, false);
    acc[0][1] = __builtin_amdgcn_wmma_f32_16x16x32_f16(false, a0, false, b1, (short)0, acc[0][1], false, false);
    acc[1][0] = __builtin_amdgcn_wmma_f32_16x16x32_f16(false, a1, false, b0, (short)0, acc[1][0], false, false);
    acc[1][1] = __builtin_amdgcn_wmma_f32_16x16x32_f16(false, a1, false, b1, (short)0, acc[1][1], false, false);
    __syncthreads();
  }

  // epilogue: bias + optional LeakyReLU, store f16
  int l = tid & 31, col0 = l & 15, rsel = l >> 4;
#pragma unroll
  for (int mi = 0; mi < 2; ++mi)
#pragma unroll
    for (int ni = 0; ni < 2; ++ni) {
      int cg = n0 + wn * 32 + ni * 16 + col0;
      float bb = bias ? bias[cg] : 0.f;
#pragma unroll
      for (int v = 0; v < 8; ++v) {
        size_t rg = m0 + wm * 32 + mi * 16 + v + 8 * rsel;
        float x = acc[mi][ni][v] + bb;
        if (act) x = leaky(x);
        out[rg * ldc + cg] = (_Float16)x;
      }
    }
}

// ---- recurrent LSTM: 32 blocks x 16 batch rows, 200 sequential WMMA steps --
// dynamic LDS: Whh [512][130] f16 | h [16][130] f16 | gates [16][512] f32
__global__ void k_lstm(const _Float16* __restrict__ xg, const _Float16* __restrict__ Whh,
                       _Float16* __restrict__ hcat, int dirofs, int t0, int dt) {
  extern __shared__ char smem[];
  _Float16* whh_s = (_Float16*)smem;                 // 512*130
  _Float16* h_s   = whh_s + 512 * 130;               // 16*130
  float*    g_s   = (float*)(h_s + 16 * 130);        // 16*512

  int tid = threadIdx.x;
  int wave = tid >> 5;
  int l = tid & 31, col0 = l & 15, rsel = l >> 4;
  int n0 = wave * 64;                                // each wave owns 64 gate cols
  int b0 = blockIdx.x * 16;

  for (int i = tid; i < 512 * 128; i += 256) {       // Whh resident in LDS (all 200 steps)
    int r = i >> 7, k = i & 127;
    whh_s[r * 130 + k] = Whh[i];
  }
  for (int i = tid; i < 16 * 130; i += 256) h_s[i] = (_Float16)0.f;
  __syncthreads();

  float c8[8];
#pragma unroll
  for (int q = 0; q < 8; ++q) c8[q] = 0.f;

  for (int s = 0; s < Ln; ++s) {
    int t = t0 + dt * s;
    // init accumulators with precomputed xg (bias already fused)
    v8f acc[4];
#pragma unroll
    for (int j = 0; j < 4; ++j) {
      int cg = n0 + j * 16 + col0;
#pragma unroll
      for (int v = 0; v < 8; ++v) {
        size_t rg = (size_t)(b0 + v + 8 * rsel) * Ln + t;
        acc[j][v] = (float)xg[rg * 512 + cg];
      }
    }
    // g += h_{t-1} @ Whh^T  : 16 WMMAs per wave per step
#pragma unroll
    for (int kk = 0; kk < 4; ++kk) {
      v16h a = frag_a(h_s + kk * 32, 130);
#pragma unroll
      for (int j = 0; j < 4; ++j) {
        v16h b = frag_b(whh_s + (size_t)(n0 + j * 16) * 130 + kk * 32, 130);
        acc[j] = __builtin_amdgcn_wmma_f32_16x16x32_f16(false, a, false, b, (short)0, acc[j], false, false);
      }
    }
    // gate pre-activations to LDS
#pragma unroll
    for (int j = 0; j < 4; ++j) {
      int cg = n0 + j * 16 + col0;
#pragma unroll
      for (int v = 0; v < 8; ++v) g_s[(v + 8 * rsel) * 512 + cg] = acc[j][v];
    }
    __syncthreads();

    // elementwise cell update: wave owns hidden cols [wave*16, wave*16+16)
    {
      int r = col0;                       // batch row 0..15
      int cbase = wave * 16 + rsel * 8;
#pragma unroll
      for (int q = 0; q < 8; ++q) {
        int j = cbase + q;
        float ig = sigm(g_s[r * 512 + j]);
        float fg = sigm(g_s[r * 512 + 128 + j]);
        float gg = tanhf(g_s[r * 512 + 256 + j]);
        float og = sigm(g_s[r * 512 + 384 + j]);
        float c = fg * c8[q] + ig * gg;
        c8[q] = c;
        float h = og * tanhf(c);
        h_s[r * 130 + j] = (_Float16)h;
        hcat[((size_t)(b0 + r) * Ln + t) * 256 + dirofs + j] = (_Float16)h;
      }
    }
    __syncthreads();
  }
}

// ---- tag projection: one wave per 16 token rows, T padded to 16 ------------
__global__ void k_logits(const _Float16* __restrict__ h2, const _Float16* __restrict__ Wtp,
                         const float* __restrict__ bt, float* __restrict__ logits) {
  size_t m0 = (size_t)blockIdx.x * 16;
  v8f acc = vzero8();
#pragma unroll
  for (int kk = 0; kk < 256; kk += 32) {
    v16h a = frag_a(h2 + m0 * 256 + kk, 256);
    v16h b = frag_b(Wtp + kk, 256);
    acc = __builtin_amdgcn_wmma_f32_16x16x32_f16(false, a, false, b, (short)0, acc, false, false);
  }
  int l = threadIdx.x & 31, col = l & 15, rsel = l >> 4;
  if (col < 6) {
    float bb = bt[col];
#pragma unroll
    for (int v = 0; v < 8; ++v) {
      size_t row = m0 + v + 8 * rsel;
      logits[row * 6 + col] = leaky(acc[v] + bb);
    }
  }
}

// ---- Viterbi: one wave per batch element, T=6 via lane shuffles ------------
__global__ void k_viterbi(const float* __restrict__ logits, const float* __restrict__ trans,
                          float* __restrict__ scores, int* __restrict__ paths) {
  __shared__ short bp[(Ln - 1) * 6];
  int b = blockIdx.x;
  int l = threadIdx.x;
  const float* lg = logits + (size_t)b * Ln * 6;
  float prev = (l < 6) ? lg[l] : -3.0e38f;
  float tr[6];
  if (l < 6) {
#pragma unroll
    for (int i = 0; i < 6; ++i) tr[i] = trans[i * 6 + l];
  }
  for (int s = 0; s < Ln - 1; ++s) {
    float best = -3.4e38f;
    int bi = 0;
#pragma unroll
    for (int i = 0; i < 6; ++i) {
      float pi = __shfl(prev, i, 32);
      float v = pi + (l < 6 ? tr[i] : 0.f);
      if (v > best) { best = v; bi = i; }   // strict > keeps first index (jnp.argmax)
    }
    if (l < 6) {
      bp[s * 6 + l] = (short)bi;
      prev = lg[(s + 1) * 6 + l] + best;
    }
  }
  __syncthreads();
  float best = -3.4e38f;
  int btag = 0;
#pragma unroll
  for (int i = 0; i < 6; ++i) {
    float pi = __shfl(prev, i, 32);
    if (pi > best) { best = pi; btag = i; }
  }
  if (l == 0) {
    scores[b] = best;
    int* path = paths + (size_t)b * Ln;
    int tag = btag;
    path[Ln - 1] = tag;
    for (int s = Ln - 2; s >= 0; --s) { tag = (int)bp[s * 6 + tag]; path[s] = tag; }
  }
}

// ---------------------------------------------------------------------------
extern "C" void kernel_launch(void* const* d_in, const int* in_sizes, int n_in,
                              void* d_out, int out_size, void* d_ws, size_t ws_size,
                              hipStream_t stream) {
  const int*   sent   = (const int*)d_in[0];
  const float* emb    = (const float*)d_in[2];
  const float* Wih_f  = (const float*)d_in[3];
  const float* Whh_f  = (const float*)d_in[4];
  const float* bih_f  = (const float*)d_in[5];
  const float* bhh_f  = (const float*)d_in[6];
  const float* Wih_b  = (const float*)d_in[7];
  const float* Whh_b  = (const float*)d_in[8];
  const float* bih_b  = (const float*)d_in[9];
  const float* bhh_b  = (const float*)d_in[10];
  const float* W1     = (const float*)d_in[11];
  const float* b1     = (const float*)d_in[12];
  const float* W2     = (const float*)d_in[13];
  const float* b2     = (const float*)d_in[14];
  const float* Wt     = (const float*)d_in[15];
  const float* bt     = (const float*)d_in[16];
  const float* trans  = (const float*)d_in[17];

  char* ws = (char*)d_ws;
  _Float16* xh     = (_Float16*)(ws + OFF_XH);
  _Float16* xg     = (_Float16*)(ws + OFF_XG);
  _Float16* hcat   = (_Float16*)(ws + OFF_HCAT);
  _Float16* wh     = (_Float16*)(ws + OFF_WH);
  float*    biasf  = (float*)(ws + OFF_BIAS);
  float*    logits = (float*)(ws + OFF_LOGITS);
  _Float16* h1     = xg;   // xg region dead after both LSTM passes
  _Float16* h2     = xh;   // x region dead after second input GEMM

  float* scores = (float*)d_out;
  int*   paths  = (int*)((float*)d_out + Bn);

  // 1) weight cast / bias fuse
  {
    int tot = W_TOTAL + 1024;
    k_prep<<<(tot + 255) / 256, 256, 0, stream>>>(Wih_f, Whh_f, bih_f, bhh_f,
                                                  Wih_b, Whh_b, bih_b, bhh_b,
                                                  W1, W2, Wt, wh, biasf);
  }
  // 2) embedding gather -> f16
  k_embed<<<M_TOT, 256, 0, stream>>>(sent, emb, xh);

  size_t lstm_lds = (size_t)(512 * 130 + 16 * 130) * sizeof(_Float16)
                  + (size_t)(16 * 512) * sizeof(float);   // 170048 B < 320 KB

  // 3) xg_f = x @ Wih_f^T + (bih_f + bhh_f)
  k_gemm<<<dim3(512 / 128, M_TOT / 64), 256, 0, stream>>>(
      xh, wh + WO_WIHF, biasf, xg, En, 512, En, 0);
  // 4) forward LSTM
  k_lstm<<<Bn / 16, 256, lstm_lds, stream>>>(xg, wh + WO_WHHF, hcat, 0, 0, 1);
  // 5) xg_b = x @ Wih_b^T + (bih_b + bhh_b)
  k_gemm<<<dim3(512 / 128, M_TOT / 64), 256, 0, stream>>>(
      xh, wh + WO_WIHB, biasf + 512, xg, En, 512, En, 0);
  // 6) backward LSTM (time-reversed)
  k_lstm<<<Bn / 16, 256, lstm_lds, stream>>>(xg, wh + WO_WHHB, hcat, Hn, Ln - 1, -1);

  // 7) h1 = leaky(hcat @ W1^T + b1)
  k_gemm<<<dim3(256 / 128, M_TOT / 64), 256, 0, stream>>>(
      hcat, wh + WO_W1, b1, h1, 256, 256, 256, 1);
  // 8) h2 = leaky(h1 @ W2^T + b2)
  k_gemm<<<dim3(256 / 128, M_TOT / 64), 256, 0, stream>>>(
      h1, wh + WO_W2, b2, h2, 256, 256, 256, 1);
  // 9) logits = leaky(h2 @ Wt^T + bt)
  k_logits<<<M_TOT / 16, 32, 0, stream>>>(h2, wh + WO_WT, bt, logits);
  // 10) Viterbi decode
  k_viterbi<<<Bn, 32, 0, stream>>>(logits, trans, scores, paths);
}
